// EdisonPerceiverAttention_14396730376855
// MI455X (gfx1250) — compile-verified, matrix operand order checked by reading, not run on hardware
//
#include <hip/hip_runtime.h>
#include <hip/hip_bf16.h>
#include <math.h>

// ---------------------------------------------------------------------------
// EdisonPerceiverAttention on MI455X (gfx1250, wave32, WMMA)
//   All GEMMs + attention via v_wmma_f32_16x16x32_f16 (f16 operands, f32 accum)
//   Every WMMA operand is fed by 2x 128-bit loads (global or LDS):
//   weights and V are pre-transposed so B-operands read contiguously.
//   Attention is flash-style (online softmax) -- materializing scores would
//   cost 2x134MB of HBM traffic vs ~170MB for the whole pipeline.
// ---------------------------------------------------------------------------

typedef __attribute__((ext_vector_type(16))) _Float16 v16h;
typedef __attribute__((ext_vector_type(8)))  _Float16 v8h;
typedef __attribute__((ext_vector_type(8)))  float    v8f;

namespace cfg {
constexpr int B = 4, N = 2048, M = 256;
constexpr int DIN = 1024, INNER = 1024, H = 16, D = 64;
constexpr float SCALE = 0.125f;      // D^-0.5
constexpr float RMS_EPS = 1e-8f;
constexpr float LN_EPS = 1e-5f;
}

__device__ __forceinline__ v8f wmma_f16(v16h a, v16h b, v8f c) {
  return __builtin_amdgcn_wmma_f32_16x16x32_f16(
      /*neg_a=*/false, a, /*neg_b=*/false, b,
      /*c_mod=*/(short)0, c, /*reuse_a=*/false, /*reuse_b=*/false);
}

// 16x32 f16 operand from a row-major tile (A-pattern; also serves as the
// B-operand when the B matrix is stored transposed, i.e. [N][K] row-major).
// ISA layout: lane r=lane&15 holds row r; lanes<16 hold K {0..7,16..23},
// lanes>=16 hold K {8..15,24..31}.  Needs 16B alignment (kh is 0 or 8 halves).
__device__ __forceinline__ v16h load_tile16x32(const _Float16* __restrict__ base,
                                               int ld, int lane) {
  const int r  = lane & 15;
  const int kh = (lane >> 4) * 8;
  const _Float16* p = base + (size_t)r * ld + kh;
  v8h lo = *(const v8h*)(p);        // -> global_load_b128 / ds_load_b128
  v8h hi = *(const v8h*)(p + 16);
  v16h a;
#pragma unroll
  for (int e = 0; e < 8; ++e) { a[e] = lo[e]; a[e + 8] = hi[e]; }
  return a;
}

// ---------------------------------------------------------------------------
// LDS-tiled transpose + f32->f16: in[Mrows][Ncols] f32 -> out[Ncols][Mrows] f16
// grid (Mrows/64, Ncols/64), block 256. Coalesced 64B reads and 32B writes.
__global__ __launch_bounds__(256)
void transpose_cvt_kernel(const float* __restrict__ in, _Float16* __restrict__ out,
                          int Mrows, int Ncols) {
  __shared__ _Float16 tile[64][65];
  const int r0 = blockIdx.x * 64, c0 = blockIdx.y * 64;
  const int t  = threadIdx.x;
  const int lr = t >> 2;            // 0..63
  const int lc = (t & 3) * 16;      // 0,16,32,48
  const float* ip = in + (size_t)(r0 + lr) * Ncols + c0 + lc;
#pragma unroll
  for (int j = 0; j < 16; ++j) tile[lc + j][lr] = (_Float16)ip[j];
  __syncthreads();
  _Float16* op = out + (size_t)(c0 + lr) * Mrows + r0 + lc;
#pragma unroll
  for (int j = 0; j < 16; ++j) op[j] = tile[lr][lc + j];
}

// v-half of kvraw -> vT[b][h][64][2048] f16 (transposed for P@V B-operands)
// grid (N/64, 1, B*H), block 256
__global__ __launch_bounds__(256)
void v_transpose_kernel(const float* __restrict__ kvraw, _Float16* __restrict__ vT) {
  __shared__ _Float16 tile[64][65];
  const int n0 = blockIdx.x * 64;
  const int bh = blockIdx.z;              // b*16 + h
  const int b = bh >> 4, h = bh & 15;
  const int t  = threadIdx.x;
  const int lr = t >> 2;
  const int lc = (t & 3) * 16;
  const float* ip = kvraw + ((size_t)b * cfg::N + n0 + lr) * 2048 + 1024 + h * 64 + lc;
#pragma unroll
  for (int j = 0; j < 16; ++j) tile[lc + j][lr] = (_Float16)ip[j];   // tile[d][n]
  __syncthreads();
  _Float16* op = vT + ((size_t)bh * 64 + lr) * (size_t)cfg::N + n0 + lc;  // row d=lr
#pragma unroll
  for (int j = 0; j < 16; ++j) op[j] = tile[lr][lc + j];
}

// ---------------------------------------------------------------------------
// LayerNorm one row per block (cols = 1024), f16 output
__global__ __launch_bounds__(256)
void layernorm_kernel(const float* __restrict__ x, const float* __restrict__ w,
                      const float* __restrict__ bb, _Float16* __restrict__ out,
                      int cols) {
  __shared__ float r1[256];
  __shared__ float r2[256];
  const int row = blockIdx.x, t = threadIdx.x;
  const float* xr = x + (size_t)row * cols;
  float s = 0.f, s2 = 0.f;
  for (int i = t; i < cols; i += 256) { float v = xr[i]; s += v; s2 += v * v; }
  r1[t] = s; r2[t] = s2; __syncthreads();
  for (int o = 128; o > 0; o >>= 1) {
    if (t < o) { r1[t] += r1[t + o]; r2[t] += r2[t + o]; }
    __syncthreads();
  }
  const float mu  = r1[0] / cols;
  const float var = r2[0] / cols - mu * mu;
  const float inv = rsqrtf(var + cfg::LN_EPS);
  _Float16* orow = out + (size_t)row * cols;
  for (int i = t; i < cols; i += 256)
    orow[i] = (_Float16)((xr[i] - mu) * inv * w[i] + bb[i]);
}

// ---------------------------------------------------------------------------
// WMMA GEMM: C[M,N] = A[M,K](f16 rm) @ BT[N,K](f16 rm, pre-transposed) (+bias)
// block = 256 threads = 8 waves; each wave owns a 16x64 output tile.
// grid = (M/16, N/512). All operand loads are 2x b128.
__global__ __launch_bounds__(256)
void gemm_wmma_kernel(const _Float16* __restrict__ A,
                      const _Float16* __restrict__ BT,
                      float* __restrict__ outF32, _Float16* __restrict__ outF16,
                      const float* __restrict__ bias, int M, int N, int K) {
  const int lane = threadIdx.x & 31;
  const int wave = threadIdx.x >> 5;
  const int rowBlk  = blockIdx.x;
  const int colBase = (blockIdx.y * 8 + wave) * 64;
  const _Float16* Arow = A + (size_t)rowBlk * 16 * K;
  const int r = lane & 15, kh = (lane >> 4) * 8;
  v8f acc[4] = {};
  for (int k0 = 0; k0 < K; k0 += 32) {
    if (k0 + 32 < K) {  // -> global_prefetch_b8 on next tiles
      __builtin_prefetch(Arow + (size_t)r * K + (k0 + 32) + kh, 0, 1);
      __builtin_prefetch(BT + (size_t)(colBase + r) * K + (k0 + 32) + kh, 0, 1);
    }
    v16h a = load_tile16x32(Arow + k0, K, lane);
#pragma unroll
    for (int t = 0; t < 4; ++t) {
      v16h b = load_tile16x32(BT + (size_t)(colBase + t * 16) * K + k0, K, lane);
      acc[t] = wmma_f16(a, b, acc[t]);
    }
  }
  // C layout: element g -> row = g + (lane>>4)*8, col = lane&15
#pragma unroll
  for (int t = 0; t < 4; ++t) {
#pragma unroll
    for (int g = 0; g < 8; ++g) {
      const int row = rowBlk * 16 + g + (lane >> 4) * 8;
      const int col = colBase + t * 16 + (lane & 15);
      float v = acc[t][g] + (bias ? bias[col] : 0.f);
      if (outF32) outF32[(size_t)row * N + col] = v;
      if (outF16) outF16[(size_t)row * N + col] = (_Float16)v;
    }
  }
}

// ---------------------------------------------------------------------------
// Per-head RMS norm of q (x gamma_q x SCALE); qraw [B*M,1024] -> qn [b][h][m][64] f16
__global__ __launch_bounds__(256)
void q_rms_kernel(const float* __restrict__ qraw, const float* __restrict__ gamma,
                  _Float16* __restrict__ qn, int rowsPerB, float scale) {
  const int row = blockIdx.x;                 // b*rowsPerB + i
  const int b = row / rowsPerB, i = row % rowsPerB;
  const int t = threadIdx.x;                  // 4 contiguous elems per thread
  const int head = t >> 4;                    // 16 threads per head (lane-half aligned)
  const int base = t * 4, d0 = base & 63;
  const float* qr = qraw + (size_t)row * 1024;
  float q0 = qr[base], q1 = qr[base + 1], q2 = qr[base + 2], q3 = qr[base + 3];
  float ss = q0 * q0 + q1 * q1 + q2 * q2 + q3 * q3;
  for (int o = 1; o < 16; o <<= 1) ss += __shfl_xor(ss, o, 32);
  const float inv = 1.0f / fmaxf(sqrtf(ss) * 0.125f, cfg::RMS_EPS);
  _Float16* op = qn + ((((size_t)b * cfg::H + head) * rowsPerB + i) * 64 + d0);
  op[0] = (_Float16)(q0 * inv * gamma[d0 + 0] * scale);
  op[1] = (_Float16)(q1 * inv * gamma[d0 + 1] * scale);
  op[2] = (_Float16)(q2 * inv * gamma[d0 + 2] * scale);
  op[3] = (_Float16)(q3 * inv * gamma[d0 + 3] * scale);
}

// k-half of kvraw: RMS*gamma_k -> kn [b][h][n][64] f16
__global__ __launch_bounds__(256)
void kv_post_kernel(const float* __restrict__ kvraw,
                    const float* __restrict__ gamma_k,
                    _Float16* __restrict__ kn, int rowsPerB) {
  const int row = blockIdx.x;
  const int b = row / rowsPerB, i = row % rowsPerB;
  const int t = threadIdx.x;
  const int head = t >> 4;
  const int base = t * 4, d0 = base & 63;
  const float* kr = kvraw + (size_t)row * 2048;
  float k0 = kr[base], k1 = kr[base + 1], k2 = kr[base + 2], k3 = kr[base + 3];
  float ss = k0 * k0 + k1 * k1 + k2 * k2 + k3 * k3;
  for (int o = 1; o < 16; o <<= 1) ss += __shfl_xor(ss, o, 32);
  const float inv = 1.0f / fmaxf(sqrtf(ss) * 0.125f, cfg::RMS_EPS);
  const size_t ob = (((size_t)b * cfg::H + head) * rowsPerB + i) * 64 + d0;
  kn[ob + 0] = (_Float16)(k0 * inv * gamma_k[d0 + 0]);
  kn[ob + 1] = (_Float16)(k1 * inv * gamma_k[d0 + 1]);
  kn[ob + 2] = (_Float16)(k2 * inv * gamma_k[d0 + 2]);
  kn[ob + 3] = (_Float16)(k3 * inv * gamma_k[d0 + 3]);
}

// ---------------------------------------------------------------------------
// Flash attention: 1 wave per (b,h,16-query-row block). grid = B*H*(M/16).
// Q:[b][h][256][64], K:[b][h][2048][64], vT:[b][h][64][2048] (all f16),
// mask byte per (b,n), may be null. o: [b][m][h*64+d] f16.
__global__ __launch_bounds__(32)
void flash_attn_kernel(const _Float16* __restrict__ qn,
                       const _Float16* __restrict__ kn,
                       const _Float16* __restrict__ vT,
                       const unsigned char* __restrict__ mask,
                       _Float16* __restrict__ o) {
  __shared__ __align__(16) _Float16 Plds[16 * 32];   // P tile staging (C->A relayout)
  const int mBlocks = cfg::M / 16;
  const int mb = blockIdx.x % mBlocks;
  const int bh = blockIdx.x / mBlocks;
  const int h = bh % cfg::H, b = bh / cfg::H;
  const int lane = threadIdx.x;
  const int c = lane & 15, rbase = (lane >> 4) * 8;

  const _Float16* Q  = qn + (((size_t)b * cfg::H + h) * cfg::M + mb * 16) * 64;
  const _Float16* K  = kn + ((size_t)b * cfg::H + h) * (size_t)cfg::N * 64;
  const _Float16* Vt = vT + ((size_t)b * cfg::H + h) * 64 * (size_t)cfg::N;
  const unsigned char* mrow = mask ? (mask + (size_t)b * cfg::N) : nullptr;

  const v16h aQ0 = load_tile16x32(Q, 64, lane);       // d 0..31
  const v16h aQ1 = load_tile16x32(Q + 32, 64, lane);  // d 32..63

  float mstat[8], lstat[8];
  v8f acc[4] = {};
#pragma unroll
  for (int g = 0; g < 8; ++g) { mstat[g] = -3.0e38f; lstat[g] = 0.f; }

  for (int j0 = 0; j0 < cfg::N; j0 += 32) {
    const _Float16* Kb = K + (size_t)j0 * 64;
    // S tiles: B-operand = K^T == A-pattern load of row-major K rows
    v8f s0 = {}, s1 = {};
    s0 = wmma_f16(aQ0, load_tile16x32(Kb, 64, lane), s0);
    s0 = wmma_f16(aQ1, load_tile16x32(Kb + 32, 64, lane), s0);
    s1 = wmma_f16(aQ0, load_tile16x32(Kb + 16 * 64, 64, lane), s1);
    s1 = wmma_f16(aQ1, load_tile16x32(Kb + 16 * 64 + 32, 64, lane), s1);

    if (mrow) {  // mask only the c1 stream (per reference)
      if (!mrow[j0 + c]) {
#pragma unroll
        for (int g = 0; g < 8; ++g) s0[g] = -3.0e38f;
      }
      if (!mrow[j0 + 16 + c]) {
#pragma unroll
        for (int g = 0; g < 8; ++g) s1[g] = -3.0e38f;
      }
    }

    // online softmax: each row lives in one 16-lane half -> shfl_xor 1,2,4,8
#pragma unroll
    for (int g = 0; g < 8; ++g) {
      float mx = fmaxf(s0[g], s1[g]);
      for (int off = 1; off < 16; off <<= 1) mx = fmaxf(mx, __shfl_xor(mx, off, 32));
      const float mnew  = fmaxf(mstat[g], mx);
      const float alpha = __expf(mstat[g] - mnew);
      const float p0 = __expf(s0[g] - mnew);
      const float p1 = __expf(s1[g] - mnew);
      float ps = p0 + p1;
      for (int off = 1; off < 16; off <<= 1) ps += __shfl_xor(ps, off, 32);
      lstat[g] = lstat[g] * alpha + ps;
      mstat[g] = mnew;
      acc[0][g] *= alpha; acc[1][g] *= alpha; acc[2][g] *= alpha; acc[3][g] *= alpha;
      s0[g] = p0; s1[g] = p1;
    }

    // stage P (16x32 f16) through LDS to convert C-layout -> A-layout
#pragma unroll
    for (int g = 0; g < 8; ++g) {
      Plds[(rbase + g) * 32 + c]      = (_Float16)s0[g];
      Plds[(rbase + g) * 32 + 16 + c] = (_Float16)s1[g];
    }
    __syncthreads();
    const v16h aP = load_tile16x32(Plds, 32, lane);
    // P@V: B-operand row = d (contiguous along keys in vT) -> 2x b128 each
#pragma unroll
    for (int t = 0; t < 4; ++t)
      acc[t] = wmma_f16(aP,
                        load_tile16x32(Vt + (size_t)(t * 16) * cfg::N + j0,
                                       cfg::N, lane),
                        acc[t]);
    __syncthreads();
  }

  // normalize and emit o[b][m][h*64+d]
#pragma unroll
  for (int g = 0; g < 8; ++g) {
    const float inv = 1.0f / lstat[g];
    const int row = mb * 16 + g + rbase;
    const size_t orow = ((size_t)b * cfg::M + row) * (size_t)cfg::INNER + (size_t)h * 64;
#pragma unroll
    for (int t = 0; t < 4; ++t)
      o[orow + t * 16 + c] = (_Float16)(acc[t][g] * inv);
  }
}

// ---------------------------------------------------------------------------
extern "C" void kernel_launch(void* const* d_in, const int* in_sizes, int n_in,
                              void* d_out, int out_size, void* d_ws, size_t ws_size,
                              hipStream_t stream) {
  using namespace cfg;
  const float* x    = (const float*)d_in[0];
  const float* lc1  = (const float*)d_in[1];
  const float* lc0  = (const float*)d_in[2];
  const unsigned char* cmask = (const unsigned char*)d_in[3];  // jax bool = 1 byte
  const float* ln_x_w = (const float*)d_in[4];
  const float* ln_x_b = (const float*)d_in[5];
  const float* ln_l_w = (const float*)d_in[6];
  const float* ln_l_b = (const float*)d_in[7];
  const float* gq   = (const float*)d_in[8];
  const float* gk   = (const float*)d_in[9];
  const float* Wq   = (const float*)d_in[10];
  const float* Wkv  = (const float*)d_in[11];
  const float* Wout = (const float*)d_in[12];
  const float* bout = (const float*)d_in[13];
  float* out = (float*)d_out;

  // scratch carve-out (~150 MB)
  char* wsp = (char*)d_ws;
  size_t off = 0;
  auto take = [&](size_t bytes) -> char* {
    char* p = wsp + off;
    off = (off + bytes + 255) & ~(size_t)255;
    return p;
  };
  _Float16* xn     = (_Float16*)take((size_t)B * N * DIN * 2);
  _Float16* l1n    = (_Float16*)take((size_t)B * M * DIN * 2);
  _Float16* l0n    = (_Float16*)take((size_t)B * M * DIN * 2);
  _Float16* WqT    = (_Float16*)take((size_t)DIN * INNER * 2);      // [INNER][DIN]
  _Float16* WkvT   = (_Float16*)take((size_t)DIN * 2 * INNER * 2);  // [2*INNER][DIN]
  _Float16* WoutT  = (_Float16*)take((size_t)INNER * DIN * 2);      // [DIN][INNER]
  float*    kvraw  = (float*)take((size_t)B * N * 2 * INNER * 4);
  float*    q1raw  = (float*)take((size_t)B * M * INNER * 4);
  float*    q0raw  = (float*)take((size_t)B * M * INNER * 4);
  _Float16* qn1    = (_Float16*)take((size_t)B * M * INNER * 2);
  _Float16* qn0    = (_Float16*)take((size_t)B * M * INNER * 2);
  _Float16* knb    = (_Float16*)take((size_t)B * N * INNER * 2);
  _Float16* vT     = (_Float16*)take((size_t)B * N * INNER * 2);
  _Float16* o1     = (_Float16*)take((size_t)B * M * INNER * 2);
  _Float16* o0     = (_Float16*)take((size_t)B * M * INNER * 2);

  // 1) weights -> f16, transposed (so GEMM B-operands are 2x b128 loads)
  transpose_cvt_kernel<<<dim3(DIN / 64, INNER / 64), 256, 0, stream>>>(Wq, WqT, DIN, INNER);
  transpose_cvt_kernel<<<dim3(DIN / 64, 2 * INNER / 64), 256, 0, stream>>>(Wkv, WkvT, DIN, 2 * INNER);
  transpose_cvt_kernel<<<dim3(INNER / 64, DIN / 64), 256, 0, stream>>>(Wout, WoutT, INNER, DIN);

  // 2) LayerNorms
  layernorm_kernel<<<B * N, 256, 0, stream>>>(x, ln_x_w, ln_x_b, xn, DIN);
  layernorm_kernel<<<B * M, 256, 0, stream>>>(lc1, ln_l_w, ln_l_b, l1n, DIN);
  layernorm_kernel<<<B * M, 256, 0, stream>>>(lc0, ln_l_w, ln_l_b, l0n, DIN);

  // 3) Projections (WMMA)
  gemm_wmma_kernel<<<dim3(B * N / 16, (2 * INNER) / 512), 256, 0, stream>>>(
      xn, WkvT, kvraw, nullptr, nullptr, B * N, 2 * INNER, DIN);
  gemm_wmma_kernel<<<dim3(B * M / 16, INNER / 512), 256, 0, stream>>>(
      l1n, WqT, q1raw, nullptr, nullptr, B * M, INNER, DIN);
  gemm_wmma_kernel<<<dim3(B * M / 16, INNER / 512), 256, 0, stream>>>(
      l0n, WqT, q0raw, nullptr, nullptr, B * M, INNER, DIN);

  // 4) Per-head RMS norms -> f16 head-major layouts; V transposed
  q_rms_kernel<<<B * M, 256, 0, stream>>>(q1raw, gq, qn1, M, SCALE);
  q_rms_kernel<<<B * M, 256, 0, stream>>>(q0raw, gq, qn0, M, SCALE);
  kv_post_kernel<<<B * N, 256, 0, stream>>>(kvraw, gk, knb, N);
  v_transpose_kernel<<<dim3(N / 64, 1, B * H), 256, 0, stream>>>(kvraw, vT);

  // 5) Flash attention (WMMA), c1 masked / c0 unmasked
  flash_attn_kernel<<<B * H * (M / 16), 32, 0, stream>>>(qn1, knb, vT, cmask, o1);
  flash_attn_kernel<<<B * H * (M / 16), 32, 0, stream>>>(qn0, knb, vT, nullptr, o0);

  // 6) Output projection (+bias), fp32 straight into d_out (tuple order: o1, o0)
  gemm_wmma_kernel<<<dim3(B * M / 16, DIN / 512), 256, 0, stream>>>(
      o1, WoutT, out, nullptr, bout, B * M, DIN, INNER);
  gemm_wmma_kernel<<<dim3(B * M / 16, DIN / 512), 256, 0, stream>>>(
      o0, WoutT, out + (size_t)B * M * DIN, nullptr, bout, B * M, DIN, INNER);
}
